// AttentionHead_33449205301357
// MI455X (gfx1250) — matrix operand order repaired and checked by reading, not compiled, not run on hardware
//
#include <hip/hip_runtime.h>

#define S_LEN  2048
#define BATCHN 4
#define DMODEL 1024
#define DINT   64
#define NTOK   (S_LEN * BATCHN)   // 8192 tokens

#ifndef __has_builtin
#define __has_builtin(x) 0
#endif
#if __has_builtin(__builtin_amdgcn_global_load_async_to_lds_b128) && \
    __has_builtin(__builtin_amdgcn_s_wait_asynccnt)
#define USE_ASYNC_LDS 1
#else
#define USE_ASYNC_LDS 0
#endif

#define AS1 __attribute__((address_space(1)))
#define AS3 __attribute__((address_space(3)))

typedef __attribute__((ext_vector_type(4)))  int    v4i;
typedef __attribute__((ext_vector_type(8)))  __bf16 v8bf;
typedef __attribute__((ext_vector_type(16))) __bf16 v16bf;
typedef __attribute__((ext_vector_type(8)))  float  v8f;

__device__ __forceinline__ v8f wmma_bf16(v16bf a, v16bf b, v8f c) {
  // (neg_a, A, neg_b, B, c_mod, C, reuse_a, reuse_b)
  return __builtin_amdgcn_wmma_f32_16x16x32_bf16(false, a, false, b, (short)0, c,
                                                 false, false);
}

__device__ __forceinline__ v16bf cat8(v8bf lo, v8bf hi) {
  return __builtin_shufflevector(lo, hi, 0, 1, 2, 3, 4, 5, 6, 7,
                                 8, 9, 10, 11, 12, 13, 14, 15);
}

// A-fragment (16x32 bf16): lane L: M=L%16, h=L/16; element e -> K=16*(e/8)+8h+(e%8)
__device__ __forceinline__ v16bf load_a_bf16(const __bf16* row, int h) {
  v8bf lo = *(const v8bf*)(row + 8 * h);
  v8bf hi = *(const v8bf*)(row + 16 + 8 * h);
  return cat8(lo, hi);
}

__device__ __forceinline__ v16bf load_a_f32(const float* row, int h) {
  v8f lo = *(const v8f*)(row + 8 * h);
  v8f hi = *(const v8f*)(row + 16 + 8 * h);
  v16bf a;
#pragma unroll
  for (int e = 0; e < 8; ++e) { a[e] = (__bf16)lo[e]; a[8 + e] = (__bf16)hi[e]; }
  return a;
}

// ---------------------------------------------------------------------------
// Projection GEMM: Y = X @ W + bias (bf16 out). Block: 8 waves, 256x64 tile,
// each wave 32 rows x 64 cols (8 WMMAs / K-step). W tile staged to LDS in
// fragment order so B-frag reads are 2 x ds_load_b128.
// TR=0 -> Y[(b*S+s)*64 + col]; TR=1 -> Y[(b*D+col)*S + s]
// ---------------------------------------------------------------------------
template <int TR>
__global__ __launch_bounds__(256)
void proj_gemm(const float* __restrict__ X, const float* __restrict__ W,
               const float* __restrict__ bias, __bf16* __restrict__ Y, int ldW) {
  __shared__ __attribute__((aligned(32))) __bf16 lw[4 * 32 * 16]; // [nt][lane][e]
  const int tid = threadIdx.x, lane = tid & 31, wv = tid >> 5;
  const int h = lane >> 4, m = lane & 15;
  const int rowBase = blockIdx.x * 256 + wv * 32;
  const int col0 = blockIdx.y * 64;

  const int s_nt = tid >> 6, s_lane = (tid >> 1) & 31, s_eh = tid & 1;
  const int s_n = col0 + s_nt * 16 + (s_lane & 15);
  const int s_k0 = 8 * (s_lane >> 4) + s_eh * 16;
  __bf16* s_dst = &lw[s_nt * 512 + s_lane * 16 + s_eh * 8];

  v8f acc[2][4] = {};
  for (int kk = 0; kk < DMODEL; kk += 32) {
    __syncthreads();
    const float* wsrc = W + (size_t)(kk + s_k0) * ldW + s_n;
    v8bf tmp;
#pragma unroll
    for (int u = 0; u < 8; ++u) tmp[u] = (__bf16)wsrc[(size_t)u * ldW];
    *(v8bf*)s_dst = tmp;                       // one ds_store_b128
    __syncthreads();

    v16bf a0 = load_a_f32(X + (size_t)(rowBase + m) * DMODEL + kk, h);
    v16bf a1 = load_a_f32(X + (size_t)(rowBase + 16 + m) * DMODEL + kk, h);
#pragma unroll
    for (int nt = 0; nt < 4; ++nt) {
      v16bf bf = *(const v16bf*)(&lw[nt * 512 + lane * 16]); // 2 x ds_load_b128
      acc[0][nt] = wmma_bf16(a0, bf, acc[0][nt]);
      acc[1][nt] = wmma_bf16(a1, bf, acc[1][nt]);
    }
  }
#pragma unroll
  for (int nt = 0; nt < 4; ++nt) {
    const int col = col0 + nt * 16 + m;
    const float bb = bias[col];
    const size_t colOff = (size_t)col * S_LEN;
#pragma unroll
    for (int rt = 0; rt < 2; ++rt) {
#pragma unroll
      for (int r = 0; r < 8; ++r) {
        const int t = rowBase + rt * 16 + r + 8 * h;  // token = s*BATCH + b
        const int b = t & 3, s = t >> 2;
        size_t idx;
        if (TR) idx = ((size_t)b << 21) + colOff + (size_t)s;             // (B,D,S)
        else    idx = ((size_t)b << 17) + ((size_t)s << 6) + (size_t)col; // (B,S,64)
        Y[idx] = (__bf16)(acc[rt][nt][r] + bb);
      }
    }
  }
}

// ---------------------------------------------------------------------------
// Scores: probs[b,i,j] = 0.125 * q . k  (K=64). Wave: 32x64 tile, no LDS.
// ---------------------------------------------------------------------------
__global__ __launch_bounds__(256)
void scores_kernel(const __bf16* __restrict__ qw, const __bf16* __restrict__ kw,
                   float* __restrict__ probs) {
  const int tid = threadIdx.x, lane = tid & 31, wv = tid >> 5;
  const int h = lane >> 4, m = lane & 15;
  const int i0 = blockIdx.x * 32;
  const int j0 = blockIdx.y * 512 + wv * 64;
  const int b  = blockIdx.z;

  const __bf16* q0 = qw + ((size_t)b * S_LEN + i0 + m) * DINT;
  const __bf16* q1 = q0 + 16 * DINT;
  v8f acc[2][4] = {};
#pragma unroll
  for (int kk = 0; kk < DINT; kk += 32) {
    v16bf a0 = load_a_bf16(q0 + kk, h);
    v16bf a1 = load_a_bf16(q1 + kk, h);
#pragma unroll
    for (int nt = 0; nt < 4; ++nt) {
      const int j = j0 + nt * 16 + m;
      v16bf bf = cat8(*(const v8bf*)(kw + ((size_t)b * S_LEN + j) * DINT + kk + 8 * h), *(const v8bf*)(kw + ((size_t)b * S_LEN + j) * DINT + kk + 16 + 8 * h));
      acc[0][nt] = wmma_bf16(a0, bf, acc[0][nt]);
      acc[1][nt] = wmma_bf16(a1, bf, acc[1][nt]);
    }
  }
#pragma unroll
  for (int nt = 0; nt < 4; ++nt) {
    const int j = j0 + nt * 16 + m;
#pragma unroll
    for (int rt = 0; rt < 2; ++rt) {
#pragma unroll
      for (int r = 0; r < 8; ++r) {
        const int i = i0 + rt * 16 + r + 8 * h;
        probs[((size_t)b << 22) + ((size_t)i << 11) + (size_t)j] = 0.125f * acc[rt][nt][r];
      }
    }
  }
}

// ---------------------------------------------------------------------------
__global__ __launch_bounds__(256)
void softmax_rows(float* __restrict__ probs) {
  __shared__ float red[256];
  float* p = probs + (size_t)blockIdx.x * S_LEN;
  const int tid = threadIdx.x;
  float v[8];
  float mx = -3.4e38f;
#pragma unroll
  for (int u = 0; u < 8; ++u) { v[u] = p[u * 256 + tid]; mx = fmaxf(mx, v[u]); }
  red[tid] = mx; __syncthreads();
  for (int s = 128; s > 0; s >>= 1) {
    if (tid < s) red[tid] = fmaxf(red[tid], red[tid + s]);
    __syncthreads();
  }
  mx = red[0]; __syncthreads();
  float sum = 0.f;
#pragma unroll
  for (int u = 0; u < 8; ++u) { v[u] = __expf(v[u] - mx); sum += v[u]; }
  red[tid] = sum; __syncthreads();
  for (int s = 128; s > 0; s >>= 1) {
    if (tid < s) red[tid] += red[tid + s];
    __syncthreads();
  }
  const float inv = 1.f / red[0];
#pragma unroll
  for (int u = 0; u < 8; ++u) p[u * 256 + tid] = v[u] * inv;
}

__global__ __launch_bounds__(256)
void colsum_kernel(const float* __restrict__ probs, float* __restrict__ cs) {
  const int b = blockIdx.y;
  const int j = blockIdx.x * 256 + threadIdx.x;
  const float* p = probs + ((size_t)b << 22) + j;
  float s = 0.f;
  for (int i = 0; i < S_LEN; ++i) s += p[(size_t)i << 11];
  cs[(b << 11) + j] = s;
}

__global__ __launch_bounds__(256)
void renorm_kernel(float* __restrict__ probs, const float* __restrict__ cs) {
  const int row = blockIdx.x;        // b*S + i
  const int b = row >> 11;
  float* p = probs + ((size_t)row << 11);
  const float* c = cs + (b << 11);
  const int tid = threadIdx.x;
#pragma unroll
  for (int u = 0; u < 8; ++u) {
    const int j = u * 256 + tid;
    p[j] = p[j] / (1e-9f + c[j]);
  }
}

// ---------------------------------------------------------------------------
// aten = probs @ V  (K=2048). V stored transposed (B,D,S). Block: 8 waves,
// 256 rows x 128 cols; the 128x32 V tile (8 KB) is staged once per K-step
// into LDS via GLOBAL_LOAD_ASYNC_TO_LDS_B128 (ASYNCcnt + barrier), shared by
// all 8 waves. Wave = 32x128 tile -> 16 WMMAs per K-step.
// ---------------------------------------------------------------------------
__global__ __launch_bounds__(256)
void attnv_kernel(const float* __restrict__ probs, const __bf16* __restrict__ vt,
                  float* __restrict__ out) {
  __shared__ __attribute__((aligned(64))) __bf16 lv[128 * 32]; // [col][k]
  const int tid = threadIdx.x, lane = tid & 31, wv = tid >> 5;
  const int h = lane >> 4, m = lane & 15;
  const int i0   = blockIdx.x * 256 + wv * 32;
  const int col0 = blockIdx.y * 128;
  const int b    = blockIdx.z;

  // staging role: thread tid copies 16 contiguous bf16 (32 B) of column s_col
  const int s_col = tid >> 1, s_k = (tid & 1) * 16;
  const __bf16* s_src0 = vt + (((size_t)b << 10) + col0 + s_col) * S_LEN + s_k;
  __bf16* s_dst0 = &lv[s_col * 32 + s_k];

  const float* p0 = probs + (((size_t)b << 11) + i0 + m) * S_LEN;
  const float* p1 = p0 + ((size_t)16 << 11);
  v8f acc[2][8] = {};
  for (int kk = 0; kk < S_LEN; kk += 32) {
    __syncthreads();               // previous tile fully consumed
#if USE_ASYNC_LDS
    __builtin_amdgcn_global_load_async_to_lds_b128(
        (AS1 v4i*)(s_src0 + kk), (AS3 v4i*)s_dst0, 0, 0);
    __builtin_amdgcn_global_load_async_to_lds_b128(
        (AS1 v4i*)(s_src0 + kk), (AS3 v4i*)s_dst0, 16, 0);
    __builtin_amdgcn_s_wait_asynccnt(0);
#else
    v8bf t0 = *(const v8bf*)(s_src0 + kk);
    v8bf t1 = *(const v8bf*)(s_src0 + kk + 8);
    *(v8bf*)s_dst0 = t0;
    *(v8bf*)(s_dst0 + 8) = t1;
#endif
    __syncthreads();               // tile visible to all waves

    if (kk + 32 < S_LEN) {
      __builtin_prefetch(p0 + kk + 32, 0, 1);
      __builtin_prefetch(p1 + kk + 32, 0, 1);
    }
    v16bf a0 = load_a_f32(p0 + kk, h);
    v16bf a1 = load_a_f32(p1 + kk, h);
#pragma unroll
    for (int nt = 0; nt < 8; ++nt) {
      const int col = nt * 16 + m;
      v16bf bf = cat8(*(const v8bf*)(&lv[col * 32 + 8 * h]), *(const v8bf*)(&lv[col * 32 + 16 + 8 * h])); // 2 x ds_load_b128
      acc[0][nt] = wmma_bf16(a0, bf, acc[0][nt]);
      acc[1][nt] = wmma_bf16(a1, bf, acc[1][nt]);
    }
  }
#pragma unroll
  for (int nt = 0; nt < 8; ++nt) {
    const int col = col0 + nt * 16 + m;
#pragma unroll
    for (int rt = 0; rt < 2; ++rt) {
#pragma unroll
      for (int r = 0; r < 8; ++r) {
        const int i = i0 + rt * 16 + r + 8 * h;
        out[(((size_t)i << 2) + b) * DMODEL + col] = acc[rt][nt][r]; // (S,B,D)
      }
    }
  }
}

// ---------------------------------------------------------------------------
extern "C" void kernel_launch(void* const* d_in, const int* in_sizes, int n_in,
                              void* d_out, int out_size, void* d_ws, size_t ws_size,
                              hipStream_t stream) {
  (void)in_sizes; (void)n_in; (void)out_size; (void)ws_size;
  const float* query = (const float*)d_in[0];
  const float* key_  = (const float*)d_in[1];
  const float* value = (const float*)d_in[2];
  const float* Wq = (const float*)d_in[3];
  const float* bq = (const float*)d_in[4];
  const float* Wk = (const float*)d_in[5];
  const float* bk = (const float*)d_in[6];
  const float* Wv = (const float*)d_in[7];
  const float* bv = (const float*)d_in[8];

  float* aten  = (float*)d_out;                                   // (S,B,D)
  float* probs = aten + (size_t)S_LEN * BATCHN * DMODEL;          // (B,S,S)

  char* ws = (char*)d_ws;
  __bf16* q_bf = (__bf16*)ws;                       // 1 MB  (B,S,64)
  __bf16* k_bf = (__bf16*)(ws + (1u << 20));        // 1 MB  (B,S,64)
  __bf16* v_t  = (__bf16*)(ws + (2u << 20));        // 16 MB (B,D,S) transposed
  float*  cs   = (float*)(ws + (18u << 20));        // 32 KB (B,S)

  dim3 blk(256);
  proj_gemm<0><<<dim3(NTOK / 256, 1),           blk, 0, stream>>>(query, Wq, bq, q_bf, DINT);
  proj_gemm<0><<<dim3(NTOK / 256, 1),           blk, 0, stream>>>(key_,  Wk, bk, k_bf, DINT);
  proj_gemm<1><<<dim3(NTOK / 256, DMODEL / 64), blk, 0, stream>>>(value, Wv, bv, v_t, DMODEL);

  scores_kernel<<<dim3(S_LEN / 32, S_LEN / 512, BATCHN), blk, 0, stream>>>(q_bf, k_bf, probs);
  softmax_rows <<<dim3(BATCHN * S_LEN), blk, 0, stream>>>(probs);
  colsum_kernel<<<dim3(S_LEN / 256, BATCHN), blk, 0, stream>>>(probs, cs);
  renorm_kernel<<<dim3(BATCHN * S_LEN), blk, 0, stream>>>(probs, cs);
  attnv_kernel <<<dim3(S_LEN / 256, DMODEL / 128, BATCHN), blk, 0, stream>>>(probs, v_t, aten);
}
